// CVCLNetwork_1065151890018
// MI455X (gfx1250) — matrix operand order repaired and checked by reading, not compiled, not run on hardware
//
#include <hip/hip_runtime.h>
#include <hip/hip_bf16.h>
#include <math.h>
#include <stdint.h>

// ---------------------------------------------------------------- constants
#define NSAMP 2048
#define DHIGH 512
#define DLOW  128
#define NCLUS 20
#define KNN   20

#define BM 128
#define BN 128
#define BK 32
#define BKP 48   // padded LDS stride (bf16): 24 dwords -> 16B-aligned fragment runs

enum { EPI_NONE = 0, EPI_BIAS = 1, EPI_BIAS_RELU = 2, EPI_BN_RELU = 3 };

typedef __attribute__((ext_vector_type(16))) __bf16 v16bf;
typedef __attribute__((ext_vector_type(8)))  float  v8f;
typedef int v4i_g __attribute__((vector_size(16)));   // matches builtin param type

#define AS1 __attribute__((address_space(1)))
#define AS3 __attribute__((address_space(3)))

#if __has_builtin(__builtin_amdgcn_global_load_async_to_lds_b128)
#define HAS_ASYNC_LDS 1
#else
#define HAS_ASYNC_LDS 0
#endif

#if HAS_ASYNC_LDS
#if __has_builtin(__builtin_amdgcn_s_wait_asynccnt)
#define WAIT_ASYNC(n) __builtin_amdgcn_s_wait_asynccnt(n)
#else
#define WAIT_ASYNC(n) asm volatile("s_wait_asynccnt %0" ::"i"(n) : "memory")
#endif
#else
#define WAIT_ASYNC(n) ((void)0)
#endif

static __device__ __forceinline__ __bf16 f2bf(float f) {
  unsigned u = __builtin_bit_cast(unsigned, f);
  unsigned r = (u + 0x7FFFu + ((u >> 16) & 1u)) >> 16;  // round-to-nearest-even
  unsigned short h = (unsigned short)r;
  return __builtin_bit_cast(__bf16, h);
}
static __device__ __forceinline__ unsigned pack2bf(float a, float b) {
  unsigned lo = __builtin_bit_cast(unsigned short, f2bf(a));
  unsigned hi = __builtin_bit_cast(unsigned short, f2bf(b));
  return lo | (hi << 16);
}

// ---------------------------------------------------------------- WMMA GEMM
// C[M,N] = A[M,K] @ B[K,N]  (both f32 row-major; bf16-converted into LDS)
__global__ __launch_bounds__(256)
void cvcl_gemm_wmma_bf16(const float* __restrict__ A, const float* __restrict__ B,
                         float* __restrict__ C, int M, int N, int K, int epi,
                         const float* __restrict__ bias, const float* __restrict__ gam,
                         const float* __restrict__ bet, const float* __restrict__ rmu,
                         const float* __restrict__ rvr)
{
  __shared__ __align__(16) __bf16 As[BM * BKP];   // [m][k]
  __shared__ __align__(16) __bf16 Bs[BN * BKP];   // [n][k]  (K innermost)

  const int tid  = threadIdx.x;
  const int lane = tid & 31;
  const int wid  = tid >> 5;
  const int wm   = wid & 3;           // 4 waves along M
  const int wn   = wid >> 2;          // 2 waves along N
  const int half = lane >> 4;
  const int l16  = lane & 15;
  const int bm0  = blockIdx.y * BM;
  const int bn0  = blockIdx.x * BN;

  v8f acc[2][4];
#pragma unroll
  for (int i = 0; i < 2; ++i)
#pragma unroll
    for (int j = 0; j < 4; ++j)
#pragma unroll
      for (int r = 0; r < 8; ++r) acc[i][j][r] = 0.f;

  const int ksteps = (K + BK - 1) / BK;
  for (int kt = 0; kt < ksteps; ++kt) {
    const int k0g = kt * BK;

    // ---- stage A tile (BM x BK): float4 global reads -> packed bf16 ds_store_b64
#pragma unroll
    for (int i = 0; i < 4; ++i) {
      int idx = tid + i * 256;        // 0..1023
      int r   = idx >> 3;             // row 0..127
      int c   = (idx & 7) << 2;       // k-col 0,4,...,28
      int gm = bm0 + r, gk = k0g + c;
      float x0 = 0.f, x1 = 0.f, x2 = 0.f, x3 = 0.f;
      if (gm < M) {
        const float* p = A + (size_t)gm * K + gk;
        if (gk + 3 < K) { float4 v = *(const float4*)p; x0 = v.x; x1 = v.y; x2 = v.z; x3 = v.w; }
        else {
          if (gk + 0 < K) x0 = p[0];
          if (gk + 1 < K) x1 = p[1];
          if (gk + 2 < K) x2 = p[2];
          if (gk + 3 < K) x3 = p[3];
        }
      }
      uint2 pk; pk.x = pack2bf(x0, x1); pk.y = pack2bf(x2, x3);
      *(uint2*)(As + r * BKP + c) = pk;
    }

    // ---- stage B tile into Bs[n][k]
#pragma unroll
    for (int i = 0; i < 4; ++i) {
      int idx = tid + i * 256;
      int kr  = idx >> 5;             // k 0..31
      int c   = (idx & 31) << 2;      // n 0,4,...,124
      int gk = k0g + kr, gn = bn0 + c;
      float x0 = 0.f, x1 = 0.f, x2 = 0.f, x3 = 0.f;
      if (gk < K) {
        const float* p = B + (size_t)gk * N + gn;
        if (gn + 3 < N) { float4 v = *(const float4*)p; x0 = v.x; x1 = v.y; x2 = v.z; x3 = v.w; }
        else {
          if (gn + 0 < N) x0 = p[0];
          if (gn + 1 < N) x1 = p[1];
          if (gn + 2 < N) x2 = p[2];
          if (gn + 3 < N) x3 = p[3];
        }
      }
      Bs[(c + 0) * BKP + kr] = f2bf(x0);
      Bs[(c + 1) * BKP + kr] = f2bf(x1);
      Bs[(c + 2) * BKP + kr] = f2bf(x2);
      Bs[(c + 3) * BKP + kr] = f2bf(x3);
    }
    __syncthreads();

    // prefetch next A tile into caches (global_prefetch_b8)
    if (kt + 1 < ksteps) {
      int pr = bm0 + (tid >> 1);
      if (pr < M && k0g + BK < K)
        __builtin_prefetch(A + (size_t)pr * K + k0g + BK, 0, 0);
    }

    // ---- fragments: two ds_load_b128 per fragment (16B-aligned runs)
    v16bf af[2]; v16bf bfr[4];
#pragma unroll
    for (int mf = 0; mf < 2; ++mf) {
      const uint4* ap = (const uint4*)(As + (wm * 32 + mf * 16 + l16) * BKP);
      uint4* au = (uint4*)&af[mf];
      au[0] = ap[half];       // dwords half*4 .. +3   (A layout: j=0..3)
      au[1] = ap[half + 2];   // dwords half*4+8 .. +11 (A layout: j=4..7)
    }
#pragma unroll
    for (int nf = 0; nf < 4; ++nf) {
      const uint4* bp = (const uint4*)(Bs + (wn * 64 + nf * 16 + l16) * BKP);
      uint4* bu = (uint4*)&bfr[nf];
      bu[0] = bp[half];           // dwords half*4 .. +3
      bu[1] = bp[half + 2];       // dwords half*4+8 .. +11
    }

#pragma unroll
    for (int mf = 0; mf < 2; ++mf)
#pragma unroll
      for (int nf = 0; nf < 4; ++nf)
        acc[mf][nf] = __builtin_amdgcn_wmma_f32_16x16x32_bf16(
            false, af[mf], false, bfr[nf], (short)0, acc[mf][nf], false, false);
    __syncthreads();
  }

  // ---- epilogue: bias / BN(eval) / ReLU fused
#pragma unroll
  for (int mf = 0; mf < 2; ++mf) {
#pragma unroll
    for (int nf = 0; nf < 4; ++nf) {
      int n = bn0 + wn * 64 + nf * 16 + l16;
#pragma unroll
      for (int r = 0; r < 8; ++r) {
        int m = bm0 + wm * 32 + mf * 16 + half * 8 + r;
        if (m < M && n < N) {
          float v = acc[mf][nf][r];
          if (epi == EPI_BIAS || epi == EPI_BIAS_RELU) {
            v += bias[n];
          } else if (epi == EPI_BN_RELU) {
            v += bias[n];
            v = (v - rmu[n]) * rsqrtf(rvr[n] + 1e-5f) * gam[n] + bet[n];
          }
          if (epi == EPI_BIAS_RELU || epi == EPI_BN_RELU) v = fmaxf(v, 0.f);
          C[(size_t)m * N + n] = v;
        }
      }
    }
  }
}

// ---------------------------------------------------------------- gram GEMM
// C[2048,2048] = Fb @ Fb^T, Fb bf16 [NSAMP, DHIGH] row-major.
// Double-buffered LDS staged with GLOBAL_LOAD_ASYNC_TO_LDS_B128 (ASYNCcnt
// partial waits; falls back to plain b128 copies if builtin unavailable).
__global__ __launch_bounds__(256)
void cvcl_gram_wmma_bf16(const __bf16* __restrict__ Fb, float* __restrict__ C)
{
  __shared__ __align__(16) __bf16 As[2][BM * BKP];
  __shared__ __align__(16) __bf16 Bs[2][BN * BKP];

  const int tid  = threadIdx.x;
  const int lane = tid & 31;
  const int wid  = tid >> 5;
  const int wm   = wid & 3;
  const int wn   = wid >> 2;
  const int half = lane >> 4;
  const int l16  = lane & 15;
  const int bm0  = blockIdx.y * BM;
  const int bn0  = blockIdx.x * BN;

  v8f acc[2][4];
#pragma unroll
  for (int i = 0; i < 2; ++i)
#pragma unroll
    for (int j = 0; j < 4; ++j)
#pragma unroll
      for (int r = 0; r < 8; ++r) acc[i][j][r] = 0.f;

  auto stage = [&](int buf, int kt) {
    const int k0 = kt * BK;
#pragma unroll
    for (int i = 0; i < 2; ++i) {
      int seg = tid + i * 256;            // 512 segments of 16B per tile
      int row = seg >> 2;                 // 0..127
      int q   = seg & 3;                  // 16B quarter within 64B row
      const __bf16* ga = Fb + (size_t)(bm0 + row) * DHIGH + k0 + q * 8;
      const __bf16* gb = Fb + (size_t)(bn0 + row) * DHIGH + k0 + q * 8;
      __bf16* la = &As[buf][row * BKP + q * 8];
      __bf16* lb = &Bs[buf][row * BKP + q * 8];
#if HAS_ASYNC_LDS
      __builtin_amdgcn_global_load_async_to_lds_b128(
          (AS1 v4i_g*)(uintptr_t)ga,
          (AS3 v4i_g*)(unsigned)(uintptr_t)la, 0, 0);
      __builtin_amdgcn_global_load_async_to_lds_b128(
          (AS1 v4i_g*)(uintptr_t)gb,
          (AS3 v4i_g*)(unsigned)(uintptr_t)lb, 0, 0);
#else
      *(uint4*)la = *(const uint4*)ga;
      *(uint4*)lb = *(const uint4*)gb;
#endif
    }
  };

  const int ksteps = DHIGH / BK;          // 16, exact
  stage(0, 0);
  int buf = 0;
  for (int kt = 0; kt < ksteps; ++kt) {
    if (kt + 1 < ksteps) {
      stage(buf ^ 1, kt + 1);
      WAIT_ASYNC(4);      // in-order completion: current tile's 4 loads retired
    } else {
      WAIT_ASYNC(0);
    }
    __syncthreads();

    v16bf af[2]; v16bf bfr[4];
#pragma unroll
    for (int mf = 0; mf < 2; ++mf) {
      const uint4* ap = (const uint4*)(&As[buf][(wm * 32 + mf * 16 + l16) * BKP]);
      uint4* au = (uint4*)&af[mf];
      au[0] = ap[half];
      au[1] = ap[half + 2];
    }
#pragma unroll
    for (int nf = 0; nf < 4; ++nf) {
      const uint4* bp = (const uint4*)(&Bs[buf][(wn * 64 + nf * 16 + l16) * BKP]);
      uint4* bu = (uint4*)&bfr[nf];
      bu[0] = bp[half];
      bu[1] = bp[half + 2];
    }
#pragma unroll
    for (int mf = 0; mf < 2; ++mf)
#pragma unroll
      for (int nf = 0; nf < 4; ++nf)
        acc[mf][nf] = __builtin_amdgcn_wmma_f32_16x16x32_bf16(
            false, af[mf], false, bfr[nf], (short)0, acc[mf][nf], false, false);
    __syncthreads();
    buf ^= 1;
  }

#pragma unroll
  for (int mf = 0; mf < 2; ++mf)
#pragma unroll
    for (int nf = 0; nf < 4; ++nf) {
      int n = bn0 + wn * 64 + nf * 16 + l16;
#pragma unroll
      for (int r = 0; r < 8; ++r) {
        int m = bm0 + wm * 32 + mf * 16 + half * 8 + r;
        __builtin_nontemporal_store(acc[mf][nf][r], &C[(size_t)m * NSAMP + n]);
      }
    }
}

// ---------------------------------------------------------------- graph kernels
__global__ void cvcl_counts(const int* __restrict__ missing, float* c1, float* c2) {
  __shared__ float s1[9], s2[9];
  if (threadIdx.x < 9) { s1[threadIdx.x] = 0.f; s2[threadIdx.x] = 0.f; }
  __syncthreads();
  float l1[9] = {0}, l2[9] = {0};
  for (int n = threadIdx.x; n < NSAMP; n += blockDim.x) {
    float p[3];
    for (int v = 0; v < 3; ++v) p[v] = (missing[n * 3 + v] == 0) ? 1.f : 0.f;
    for (int a = 0; a < 3; ++a)
      for (int b = 0; b < 3; ++b) {
        l1[a * 3 + b] += p[a] * p[b];
        l2[a * 3 + b] += p[a] * (1.f - p[b]);
      }
  }
  for (int i = 0; i < 9; ++i) { atomicAdd(&s1[i], l1[i]); atomicAdd(&s2[i], l2[i]); }
  __syncthreads();
  if (threadIdx.x < 9) { c1[threadIdx.x] = s1[threadIdx.x]; c2[threadIdx.x] = s2[threadIdx.x]; }
}

__global__ void cvcl_vecsums(const int* __restrict__ missing, const float* __restrict__ F,
                             float* __restrict__ S1, float* __restrict__ S2) {
  int p = blockIdx.x, a = p / 3, b = p % 3, d = threadIdx.x;
  const float* Fa = F + (size_t)a * NSAMP * DHIGH;
  float s1 = 0.f, s2 = 0.f;
  for (int n = 0; n < NSAMP; ++n) {
    float pa = (missing[n * 3 + a] == 0) ? 1.f : 0.f;
    float pb = (missing[n * 3 + b] == 0) ? 1.f : 0.f;
    float f  = Fa[(size_t)n * DHIGH + d];
    s1 += pa * pb * f;
    s2 += pa * (1.f - pb) * f;
  }
  S1[p * DHIGH + d] = s1;
  S2[p * DHIGH + d] = s2;
}

__global__ void cvcl_buildG(const float* S1, const float* S2,
                            const float* c1, const float* c2, float* G) {
  if (threadIdx.x || blockIdx.x) return;
  for (int a = 0; a < 3; ++a)
    for (int b = 0; b < 3; ++b) {
      float nov = c1[a * 3 + b], nau = c2[a * 3 + b], nbu = c2[b * 3 + a];
      float d1 = 0.f, d2 = 0.f;
      for (int d = 0; d < DHIGH; ++d) {
        d1 += S1[(a * 3 + b) * DHIGH + d] * S1[(b * 3 + a) * DHIGH + d];
        d2 += S2[(a * 3 + b) * DHIGH + d] * S2[(b * 3 + a) * DHIGH + d];
      }
      float simov = (nov > 0.f) ? d1 / (2.f * fmaxf(nov, 1.f)) : 0.f;
      float simu  = (nau > 0.f && nbu > 0.f) ? d2 / fmaxf(nau + nbu, 1.f) : 0.f;
      float tot = nov + nau + nbu;
      G[a * 3 + b] = (tot > 0.f) ? (simov * nov + simu * (nau + nbu)) / fmaxf(tot, 1.f) : 0.f;
    }
  float dia[3];
  for (int a = 0; a < 3; ++a) dia[a] = G[a * 3 + a];
  for (int a = 0; a < 3; ++a)
    for (int b = 0; b < 3; ++b) {
      float g = G[a * 3 + b];
      if (dia[a] != 0.f) g /= dia[a];
      if (a == b) g = 0.f;
      G[a * 3 + b] = g;
    }
}

__global__ void cvcl_scores(const float* __restrict__ G, const int* __restrict__ missing,
                            int vi, int* __restrict__ mix, float* __restrict__ wv) {
  int n = blockIdx.x * blockDim.x + threadIdx.x;
  if (n >= NSAMP) return;
  float best = -1e30f; int bi = 0;
  for (int b = 0; b < 3; ++b) {
    float s = G[vi * 3 + b] * ((missing[n * 3 + b] == 0) ? 1.f : 0.f);
    if (s > best) { best = s; bi = b; }   // first max wins (jnp.argmax)
  }
  mix[n] = bi;
  wv[n] = G[vi * 3 + bi];
}

__global__ void cvcl_rownorm(const float* __restrict__ Fv, __bf16* __restrict__ Fh) {
  int n = blockIdx.x * blockDim.x + threadIdx.x;
  if (n >= NSAMP) return;
  float s = 0.f;
  for (int d = 0; d < DHIGH; ++d) { float x = Fv[(size_t)n * DHIGH + d]; s += x * x; }
  float inv = 1.f / fmaxf(sqrtf(s), 1e-8f);
  for (int d = 0; d < DHIGH; ++d) Fh[(size_t)n * DHIGH + d] = f2bf(Fv[(size_t)n * DHIGH + d] * inv);
}

__global__ void cvcl_gather(const float* __restrict__ gram, float* __restrict__ sims,
                            const int* __restrict__ mix, int v) {
  int n = blockIdx.x;
  if (mix[n] != v) return;
  for (int c = threadIdx.x; c < NSAMP; c += blockDim.x)
    sims[(size_t)n * NSAMP + c] = gram[(size_t)n * NSAMP + c];
}

__global__ void cvcl_topk(const float* __restrict__ sims, const int* __restrict__ missing,
                          const float* __restrict__ wv, int vi,
                          float* __restrict__ wK, int* __restrict__ tiK,
                          int* __restrict__ upd, float* __restrict__ coefA) {
  int n = blockIdx.x * blockDim.x + threadIdx.x;
  if (n >= NSAMP) return;
  float tv[KNN]; int ti[KNN];
  for (int k = 0; k < KNN; ++k) { tv[k] = -1e30f; ti[k] = 0; }
  const float* row = sims + (size_t)n * NSAMP;
  for (int m = 0; m < NSAMP; ++m) {
    float v = row[m];
    if (v > tv[KNN - 1]) {
      int p = KNN - 1;
      while (p > 0 && tv[p - 1] < v) { tv[p] = tv[p - 1]; ti[p] = ti[p - 1]; --p; }
      tv[p] = v; ti[p] = m;
    }
  }
  float denom = 0.f; int anyv = 0;
  for (int k = 0; k < KNN; ++k) {
    float vv = fmaxf(tv[k], 0.f);
    int valid = (missing[ti[k] * 3 + vi] == 0);
    float w = valid ? vv : 0.f;
    wK[(size_t)n * KNN + k] = w;
    tiK[(size_t)n * KNN + k] = ti[k];
    denom += w;
    anyv |= valid;
  }
  int missn = (missing[n * 3 + vi] != 0);
  upd[n] = (missn && anyv) ? 1 : 0;
  coefA[n] = wv[n] / ((denom > 0.f) ? denom : 1.f);
}

__global__ void cvcl_impute(float* __restrict__ X, int cols,
                            const float* __restrict__ wK, const int* __restrict__ tiK,
                            const int* __restrict__ upd, const float* __restrict__ coefA) {
  int n = blockIdx.x;
  if (!upd[n]) return;                    // uniform per block
  __shared__ float sw[KNN]; __shared__ int si[KNN];
  if (threadIdx.x < KNN) {
    sw[threadIdx.x] = wK[(size_t)n * KNN + threadIdx.x];
    si[threadIdx.x] = tiK[(size_t)n * KNN + threadIdx.x];
  }
  __syncthreads();
  float cf = coefA[n];
  for (int c = threadIdx.x; c < cols; c += blockDim.x) {
    float s = 0.f;
    for (int k = 0; k < KNN; ++k) s += sw[k] * X[(size_t)si[k] * cols + c];
    X[(size_t)n * cols + c] = cf * s;
  }
}

__global__ void cvcl_label2(const float* __restrict__ H, const float* __restrict__ W2,
                            const float* __restrict__ b2, float* __restrict__ lbps, int vi) {
  int n = blockIdx.x * blockDim.x + threadIdx.x;
  if (n >= NSAMP) return;
  float lg[NCLUS];
  for (int c = 0; c < NCLUS; ++c) lg[c] = b2[c];
  for (int k = 0; k < DLOW; ++k) {
    float h = H[(size_t)n * DLOW + k];
#pragma unroll
    for (int c = 0; c < NCLUS; ++c) lg[c] += h * W2[k * NCLUS + c];
  }
  float mx = lg[0];
  for (int c = 1; c < NCLUS; ++c) mx = fmaxf(mx, lg[c]);
  float s = 0.f;
  for (int c = 0; c < NCLUS; ++c) { lg[c] = expf(lg[c] - mx); s += lg[c]; }
  float inv = 1.f / s;
  float* o = lbps + ((size_t)vi * NSAMP + n) * NCLUS;
  for (int c = 0; c < NCLUS; ++c) o[c] = lg[c] * inv;
}

__global__ void cvcl_fused(const float* __restrict__ F, float* __restrict__ fused) {
  int i = blockIdx.x * blockDim.x + threadIdx.x;
  if (i >= NSAMP * DHIGH) return;
  fused[i] = (F[i] + F[i + NSAMP * DHIGH] + F[i + 2 * NSAMP * DHIGH]) * (1.f / 3.f);
}

__global__ void cvcl_zero2(float* a, float* b) {
  if (!threadIdx.x && !blockIdx.x) { *a = 0.f; *b = 0.f; }
}

__global__ void cvcl_mse(const float* __restrict__ X, const float* __restrict__ Y,
                         long n, float scale, float* __restrict__ out) {
  __shared__ float sh[256];
  long i = (long)blockIdx.x * blockDim.x + threadIdx.x;
  long stride = (long)gridDim.x * blockDim.x;
  float a = 0.f;
  for (long j = i; j < n; j += stride) { float d = X[j] - Y[j]; a += d * d; }
  sh[threadIdx.x] = a;
  __syncthreads();
  for (int s = 128; s > 0; s >>= 1) {
    if (threadIdx.x < s) sh[threadIdx.x] += sh[threadIdx.x + s];
    __syncthreads();
  }
  if (!threadIdx.x) atomicAdd(out, sh[0] * scale);
}

// ---------------------------------------------------------------- host
extern "C" void kernel_launch(void* const* d_in, const int* in_sizes, int n_in,
                              void* d_out, int out_size, void* d_ws, size_t ws_size,
                              hipStream_t stream) {
  (void)in_sizes; (void)n_in; (void)out_size; (void)ws_size;
  static const int INS[3] = {2048, 1024, 784};

  const float* dv[3] = {(const float*)d_in[0], (const float*)d_in[1], (const float*)d_in[2]};
  const int* missing = (const int*)d_in[3];
  // d_in[4] = phase_code (setup uses 1)

  struct Lay { const float *W, *b, *beta, *g, *rm, *rv; };
  Lay enc[3][3], dec[3][3];
  int ix = 5;
  // leaf order per dict (sorted keys): W, b, [beta, g, rm, rv]
  for (int v = 0; v < 3; ++v)
    for (int l = 0; l < 3; ++l) {
      Lay L;
      L.W = (const float*)d_in[ix++]; L.b = (const float*)d_in[ix++];
      if (l < 2) {
        L.beta = (const float*)d_in[ix++]; L.g = (const float*)d_in[ix++];
        L.rm = (const float*)d_in[ix++];   L.rv = (const float*)d_in[ix++];
      } else { L.beta = L.g = L.rm = L.rv = nullptr; }
      enc[v][l] = L;
    }
  for (int v = 0; v < 3; ++v)
    for (int l = 0; l < 3; ++l) {
      Lay L;
      L.W = (const float*)d_in[ix++]; L.b = (const float*)d_in[ix++];
      if (l < 2) {
        L.beta = (const float*)d_in[ix++]; L.g = (const float*)d_in[ix++];
        L.rm = (const float*)d_in[ix++];   L.rv = (const float*)d_in[ix++];
      } else { L.beta = L.g = L.rm = L.rv = nullptr; }
      dec[v][l] = L;
    }
  const float* W1 = (const float*)d_in[ix++]; const float* b1 = (const float*)d_in[ix++];
  const float* W2 = (const float*)d_in[ix++]; const float* b2 = (const float*)d_in[ix++];

  // ---- output layout (flat, return order)
  float* out = (float*)d_out;
  float* o_lbps = out;                                   // [3,2048,20]
  float* o_rec[3];
  o_rec[0] = o_lbps + 3 * NSAMP * NCLUS;                 // [2048,2048]
  o_rec[1] = o_rec[0] + (size_t)NSAMP * 2048;            // [2048,1024]
  o_rec[2] = o_rec[1] + (size_t)NSAMP * 1024;            // [2048,784]
  float* o_fused = o_rec[2] + (size_t)NSAMP * 784;       // [2048,512]
  float* o_F = o_fused + (size_t)NSAMP * DHIGH;          // [3,2048,512]
  float* o_il = o_F + (size_t)3 * NSAMP * DHIGH;
  float* o_ol = o_il + 1;

  // ---- workspace layout
  float* ws = (float*)d_ws;
  size_t off = 0;
  float* dvn[3];
  dvn[0] = ws + off; off += (size_t)NSAMP * 2048;
  dvn[1] = ws + off; off += (size_t)NSAMP * 1024;
  dvn[2] = ws + off; off += (size_t)NSAMP * 784;
  float* act0 = ws + off; off += (size_t)NSAMP * 1024;
  float* act1 = ws + off; off += (size_t)NSAMP * 1024;
  float* tmp  = ws + off; off += (size_t)NSAMP * NSAMP;   // gram / dec_out
  float* sims = ws + off; off += (size_t)NSAMP * NSAMP;
  float* encout = ws + off; off += (size_t)NSAMP * DHIGH; // enc_out for in_loss
  __bf16* fhatb = (__bf16*)(ws + off); off += (size_t)NSAMP * DHIGH / 2;  // bf16 Fhat
  float* Hbuf = ws + off; off += (size_t)NSAMP * DLOW;
  float* S1 = ws + off; off += 9 * DHIGH;
  float* S2 = ws + off; off += 9 * DHIGH;
  float* C1 = ws + off; off += 16;
  float* C2 = ws + off; off += 16;
  float* Gm = ws + off; off += 16;
  float* wv   = ws + off; off += NSAMP;
  float* coef = ws + off; off += NSAMP;
  float* wK   = ws + off; off += (size_t)NSAMP * KNN;
  int* tiK = (int*)(ws + off); off += (size_t)NSAMP * KNN;
  int* mix = (int*)(ws + off); off += NSAMP;
  int* upd = (int*)(ws + off); off += NSAMP;

  auto gemm = [&](const float* A, const float* B, float* Cp, int M, int N, int K,
                  int epi, const float* b, const float* g, const float* be,
                  const float* rm, const float* rv) {
    dim3 gr((N + BN - 1) / BN, (M + BM - 1) / BM);
    cvcl_gemm_wmma_bf16<<<gr, 256, 0, stream>>>(A, B, Cp, M, N, K, epi, b, g, be, rm, rv);
  };
  auto mlp = [&](const float* X, Lay* P, int din, int dlast, float* outp) {
    gemm(X, P[0].W, act0, NSAMP, 1024, din, EPI_BN_RELU, P[0].b, P[0].g, P[0].beta, P[0].rm, P[0].rv);
    gemm(act0, P[1].W, act1, NSAMP, 1024, 1024, EPI_BN_RELU, P[1].b, P[1].g, P[1].beta, P[1].rm, P[1].rv);
    gemm(act1, P[2].W, outp, NSAMP, dlast, 1024, EPI_BIAS_RELU, P[2].b, 0, 0, 0, 0);
  };

  // 1) encoders -> F (in d_out, imputed in place later)
  for (int v = 0; v < 3; ++v) mlp(dv[v], enc[v], INS[v], DHIGH, o_F + (size_t)v * NSAMP * DHIGH);
  // 2) decoders -> rec (in d_out)
  for (int v = 0; v < 3; ++v) mlp(o_F + (size_t)v * NSAMP * DHIGH, dec[v], DHIGH, INS[v], o_rec[v]);
  // 3) dvn = copies of data views
  for (int v = 0; v < 3; ++v)
    (void)hipMemcpyAsync(dvn[v], dv[v], sizeof(float) * (size_t)NSAMP * INS[v],
                         hipMemcpyDeviceToDevice, stream);
  // 4) similarity graph G (from pre-imputation F)
  cvcl_counts<<<1, 256, 0, stream>>>(missing, C1, C2);
  cvcl_vecsums<<<9, 512, 0, stream>>>(missing, o_F, S1, S2);
  cvcl_buildG<<<1, 1, 0, stream>>>(S1, S2, C1, C2, Gm);

  // 5) per-view imputation loop
  for (int vi = 0; vi < 3; ++vi) {
    cvcl_scores<<<8, 256, 0, stream>>>(Gm, missing, vi, mix, wv);
    for (int v = 0; v < 3; ++v) {
      cvcl_rownorm<<<8, 256, 0, stream>>>(o_F + (size_t)v * NSAMP * DHIGH, fhatb);
      cvcl_gram_wmma_bf16<<<dim3(NSAMP / BN, NSAMP / BM), 256, 0, stream>>>(fhatb, tmp);
      cvcl_gather<<<NSAMP, 256, 0, stream>>>(tmp, sims, mix, v);
    }
    cvcl_topk<<<8, 256, 0, stream>>>(sims, missing, wv, vi, wK, tiK, upd, coef);
    cvcl_impute<<<NSAMP, 256, 0, stream>>>(o_F + (size_t)vi * NSAMP * DHIGH, DHIGH, wK, tiK, upd, coef);
    cvcl_impute<<<NSAMP, 256, 0, stream>>>(dvn[vi], INS[vi], wK, tiK, upd, coef);
    cvcl_impute<<<NSAMP, 256, 0, stream>>>(o_rec[vi], INS[vi], wK, tiK, upd, coef);
    gemm(o_F + (size_t)vi * NSAMP * DHIGH, W1, Hbuf, NSAMP, DLOW, DHIGH, EPI_BIAS, b1, 0, 0, 0, 0);
    cvcl_label2<<<8, 256, 0, stream>>>(Hbuf, W2, b2, o_lbps, vi);
  }

  // 6) fused mean
  cvcl_fused<<<(NSAMP * DHIGH + 255) / 256, 256, 0, stream>>>(o_F, o_fused);

  // 7) losses
  cvcl_zero2<<<1, 1, 0, stream>>>(o_il, o_ol);
  for (int v = 0; v < 3; ++v) {
    mlp(dvn[v], enc[v], INS[v], DHIGH, encout);
    cvcl_mse<<<256, 256, 0, stream>>>(encout, o_F + (size_t)v * NSAMP * DHIGH,
                                      (long)NSAMP * DHIGH, 1.f / ((float)NSAMP * DHIGH), o_il);
  }
  for (int v = 0; v < 3; ++v) {
    mlp(o_F + (size_t)v * NSAMP * DHIGH, dec[v], DHIGH, INS[v], tmp);
    cvcl_mse<<<256, 256, 0, stream>>>(tmp, o_rec[v],
                                      (long)NSAMP * INS[v], 1.f / ((float)NSAMP * INS[v]), o_ol);
  }
}
